// Templ_emb_50190987821737
// MI455X (gfx1250) — matrix operand order repaired and checked by reading, not compiled, not run on hardware
//
#include <hip/hip_runtime.h>
#include <hip/hip_bf16.h>
#include <stdint.h>

// ---------------------------------------------------------------------------
// CDNA5 (gfx1250) implementation of the RF2-style template embedding stack.
// Wave32; matrix math via v_wmma_f32_16x16x32_f16; global->LDS staging via
// gfx1250 async-tensor path (global_load_async_to_lds_b128 / s_wait_asynccnt).
// ---------------------------------------------------------------------------

typedef __attribute__((ext_vector_type(16))) _Float16 v16h;
typedef __attribute__((ext_vector_type(8)))  float    v8f;

union U16h { v16h v; _Float16 h[16]; };

#define LL 256              // sequence length L
#define NTPL 4              // N = B*T templates
#define SZPOS 262144        // N*L*L
#define DT 64               // d_templ
#define SCALE_Q 0.25f       // 1/sqrt(16)

// ---- gfx1250 async global->LDS copy (ASYNCcnt-tracked) --------------------
__device__ __forceinline__ void async_copy_b128(void* lds, const void* gaddr) {
  unsigned lo = (unsigned)(uintptr_t)lds;      // low 32 bits == LDS offset
  asm volatile("global_load_async_to_lds_b128 %0, %1, off"
               :: "v"(lo), "v"(gaddr) : "memory");
}
__device__ __forceinline__ void async_wait0() {
  asm volatile("s_wait_asynccnt 0" ::: "memory");
}

// ---- WMMA fragment loaders (contiguous k = 16*hi + s, same in A and B) -----------------
// A (16x32 f16): lane half selects K{0..15} vs K{16..31}.
template<int LDA>
__device__ __forceinline__ v16h a_frag_f32(float (*As)[LDA], int row0, int lm, int hi) {
  U16h u;
  int m = row0 + lm;
#pragma unroll
  for (int p = 0; p < 8; ++p) {
    int k0 = 2 * p + hi * 16;
    u.h[2 * p]     = (_Float16)As[m][k0];
    u.h[2 * p + 1] = (_Float16)As[m][k0 + 1];
  }
  return u.v;
}
__device__ __forceinline__ v16h a_frag_f16(_Float16 (*As)[32], int row0, int lm, int hi) {
  U16h u;
  int m = row0 + lm;
#pragma unroll
  for (int p = 0; p < 8; ++p) {
    int k0 = 2 * p + hi * 16;
    u.h[2 * p] = As[m][k0]; u.h[2 * p + 1] = As[m][k0 + 1];
  }
  return u.v;
}
// B (32x16 f16), LDS tile stored natural [k][n]: h[t] = B[hi*16+t][n]
template<int LDB>
__device__ __forceinline__ v16h b_frag_f32_col(float (*Bs)[LDB], int col0, int lm, int hi) {
  U16h u;
#pragma unroll
  for (int t = 0; t < 16; ++t) u.h[t] = (_Float16)Bs[hi * 16 + t][col0 + lm];
  return u.v;
}
// B tile stored transposed [n][k] with k contiguous: h[t] = B[n][hi*16+t]
template<int LDB>
__device__ __forceinline__ v16h b_frag_f32_row(float (*Bs)[LDB], int col0, int lm, int hi) {
  U16h u;
#pragma unroll
  for (int t = 0; t < 16; ++t) u.h[t] = (_Float16)Bs[col0 + lm][hi * 16 + t];
  return u.v;
}

// ---------------------------------------------------------------------------
// Generic tiled GEMM: C[M,N] = op(alpha * A[M,K]@B[K,N] + bias), optional acc.
// NT = number of 16-wide N subtiles (compile-time -> branch-free WMMA loop).
// ---------------------------------------------------------------------------
template<int NT>
__global__ __launch_bounds__(128) void k_gemm_t(
    const float* __restrict__ A, const float* __restrict__ B,
    const float* __restrict__ bias, float* __restrict__ C,
    int M, int K, int Nd, float alpha, int act, int acc)
{
  __shared__ float Asf[64][36];   // row stride 144B: 16B aligned, conflict-free
  __shared__ float Bsf[32][68];   // natural [k][n], stride 272B
  const int tid = threadIdx.x, lane = tid & 31, w = tid >> 5;
  const int lm = lane & 15, hi = lane >> 4;
  const int mbase = blockIdx.x * 64;
  const int nbase = blockIdx.y * 64;

  v8f accv[NT] = {};
  const int ksteps = (K + 31) >> 5;
  const bool fullM = (mbase + 64 <= M) && ((K & 3) == 0);
  const bool fullN = (nbase + 64 <= Nd);

  for (int ks = 0; ks < ksteps; ++ks) {
    const int kb = ks << 5;
    const bool fullK = (kb + 32 <= K);
    const bool asyA = fullM && fullK;
    const bool asyB = fullN && fullK;
    if (asyA) {
      for (int e = tid; e < 512; e += 128) {          // 4 b128 per thread
        int r = e >> 3, c4 = (e & 7) * 4;
        async_copy_b128(&Asf[r][c4], &A[(size_t)(mbase + r) * K + kb + c4]);
      }
    } else {
      for (int e = tid; e < 2048; e += 128) {
        int r = e >> 5, kk = e & 31;
        int gm = mbase + r, gk = kb + kk;
        Asf[r][kk] = (gm < M && gk < K) ? A[(size_t)gm * K + gk] : 0.f;
      }
      int pr = mbase + (tid & 63);
      if (pr < M && kb + 32 < K) __builtin_prefetch(&A[(size_t)pr * K + kb + 32]);
    }
    if (asyB) {
      for (int e = tid; e < 512; e += 128) {
        int r = e >> 4, c4 = (e & 15) * 4;
        async_copy_b128(&Bsf[r][c4], &B[(size_t)(kb + r) * Nd + nbase + c4]);
      }
    } else {
      for (int e = tid; e < 2048; e += 128) {
        int kk = e >> 6, nn = e & 63;
        int gn = nbase + nn, gk = kb + kk;
        Bsf[kk][nn] = (gn < Nd && gk < K) ? B[(size_t)gk * Nd + gn] : 0.f;
      }
    }
    if (asyA || asyB) async_wait0();
    __syncthreads();
    v16h af = a_frag_f32<36>(Asf, w << 4, lm, hi);
#pragma unroll
    for (int nt = 0; nt < NT; ++nt) {
      v16h bf = b_frag_f32_col<68>(Bsf, nt * 16, lm, hi);
      accv[nt] = __builtin_amdgcn_wmma_f32_16x16x32_f16(
          false, af, false, bf, (short)0, accv[nt], false, false);
    }
    __syncthreads();
  }

#pragma unroll
  for (int nt = 0; nt < NT; ++nt) {
#pragma unroll
    for (int r = 0; r < 8; ++r) {
      int row = mbase + (w << 4) + (hi << 3) + r;
      int col = nbase + nt * 16 + lm;
      if (row < M && col < Nd) {
        float vv = alpha * accv[nt][r];
        if (bias) vv += bias[col];
        if (act == 1) vv = fmaxf(vv, 0.f);
        else if (act == 2) vv = 1.f / (1.f + __expf(-vv));
        size_t o = (size_t)row * Nd + col;
        if (acc) C[o] += vv; else C[o] = vv;
      }
    }
  }
}

// ---------------------------------------------------------------------------
// Tied axial attention logits:
//   S[n,h,i,j] = sum_{m,d} q[pos(n,m,i),h*16+d] * k[pos(n,m,j),h*16+d] + b
// pos(n,u,a) = row_mode ? (n*L + a)*L + u : (n*L + u)*L + a   (transpose folded)
// Grid: (4 i-tiles, 4 j-tiles, n*H).  K = L*DH = 4096; all tiles full.
// ---------------------------------------------------------------------------
__global__ __launch_bounds__(128) void k_attn_logits(
    const float* __restrict__ q, const float* __restrict__ k,
    const float* __restrict__ bproj, float* __restrict__ S, int row_mode)
{
  __shared__ float Asf[64][36];
  __shared__ float Bsf[64][36];   // transposed [j][k], k contiguous
  const int tid = threadIdx.x, lane = tid & 31, w = tid >> 5;
  const int lm = lane & 15, hi = lane >> 4;
  const int ib = blockIdx.x * 64, jb = blockIdx.y * 64;
  const int nh = blockIdx.z, n = nh >> 2, h = nh & 3;

  v8f acc[4] = {};
  for (int ks = 0; ks < 128; ++ks) {
    const int m0 = ks * 2;
    for (int e = tid; e < 512; e += 128) {
      int r = e >> 3, q4 = (e & 7) * 4;
      int m = m0 + (q4 >> 4), d = q4 & 15;
      int i = ib + r;
      size_t pos = row_mode ? ((size_t)(n * LL + i) * LL + m)
                            : ((size_t)(n * LL + m) * LL + i);
      async_copy_b128(&Asf[r][q4], &q[pos * DT + h * 16 + d]);
    }
    for (int e = tid; e < 512; e += 128) {
      int c = e >> 3, q4 = (e & 7) * 4;
      int m = m0 + (q4 >> 4), d = q4 & 15;
      int j = jb + c;
      size_t pos = row_mode ? ((size_t)(n * LL + j) * LL + m)
                            : ((size_t)(n * LL + m) * LL + j);
      async_copy_b128(&Bsf[c][q4], &k[pos * DT + h * 16 + d]);
    }
    async_wait0();
    __syncthreads();
    v16h af = a_frag_f32<36>(Asf, w << 4, lm, hi);
#pragma unroll
    for (int nt = 0; nt < 4; ++nt) {
      v16h bf = b_frag_f32_row<36>(Bsf, nt * 16, lm, hi);
      acc[nt] = __builtin_amdgcn_wmma_f32_16x16x32_f16(
          false, af, false, bf, (short)0, acc[nt], false, false);
    }
    __syncthreads();
  }

#pragma unroll
  for (int nt = 0; nt < 4; ++nt) {
#pragma unroll
    for (int r = 0; r < 8; ++r) {
      int i = ib + (w << 4) + (hi << 3) + r;
      int j = jb + nt * 16 + lm;
      size_t pos = row_mode ? ((size_t)(n * LL + j) * LL + i)
                            : ((size_t)(n * LL + i) * LL + j);
      S[((size_t)nh * LL + i) * LL + j] = acc[nt][r] + bproj[pos * 4 + h];
    }
  }
}

// ---------------------------------------------------------------------------
// Attention apply: O[pos(n,f,i), h*16+d] = sum_j S[n,h,i,j] * v[pos(n,f,j), h*16+d]
// Grid: (4 i-tiles, f = 0..255, n*H).  K = 256, N = 16; all tiles full.
// ---------------------------------------------------------------------------
__global__ __launch_bounds__(128) void k_attn_out(
    const float* __restrict__ S, const float* __restrict__ v,
    float* __restrict__ O, int row_mode)
{
  __shared__ float Asf[64][36];
  __shared__ float Bsf[32][20];   // natural [k][d]
  const int tid = threadIdx.x, lane = tid & 31, w = tid >> 5;
  const int lm = lane & 15, hi = lane >> 4;
  const int ib = blockIdx.x * 64;
  const int f = blockIdx.y;
  const int nh = blockIdx.z, n = nh >> 2, h = nh & 3;

  v8f acc = {};
  for (int ks = 0; ks < 8; ++ks) {
    const int kb = ks * 32;
    for (int e = tid; e < 512; e += 128) {
      int r = e >> 3, c4 = (e & 7) * 4;
      async_copy_b128(&Asf[r][c4],
                      &S[((size_t)nh * LL + (ib + r)) * LL + kb + c4]);
    }
    for (int e = tid; e < 128; e += 128) {
      int r = e >> 2, d4 = (e & 3) * 4;
      int j = kb + r;
      size_t pos = row_mode ? ((size_t)(n * LL + j) * LL + f)
                            : ((size_t)(n * LL + f) * LL + j);
      async_copy_b128(&Bsf[r][d4], &v[pos * DT + h * 16 + d4]);
    }
    async_wait0();
    __syncthreads();
    v16h af = a_frag_f32<36>(Asf, w << 4, lm, hi);
    v16h bf = b_frag_f32_col<20>(Bsf, 0, lm, hi);
    acc = __builtin_amdgcn_wmma_f32_16x16x32_f16(
        false, af, false, bf, (short)0, acc, false, false);
    __syncthreads();
  }
#pragma unroll
  for (int r = 0; r < 8; ++r) {
    int i = ib + (w << 4) + (hi << 3) + r;
    size_t pos = row_mode ? ((size_t)(n * LL + i) * LL + f)
                          : ((size_t)(n * LL + f) * LL + i);
    O[pos * DT + h * 16 + lm] = acc[r];
  }
}

// ---------------------------------------------------------------------------
// Fused gate bias:  bias[pos,c] = sigmoid( outer(left,right)[pos,:] @ gate_w
//                                           + gate_b[c] ) * rbfW[pos,c]
// K = 256; A tile is the 16x16 outer product computed on the fly (VALU->f16),
// B tile (gate_w) staged via async copy.
// ---------------------------------------------------------------------------
__global__ __launch_bounds__(128) void k_gate_bias(
    const float* __restrict__ left, const float* __restrict__ right,
    const float* __restrict__ gate_w, const float* __restrict__ gate_b,
    const float* __restrict__ rbfW, float* __restrict__ biasOut)
{
  __shared__ _Float16 As[64][32];
  __shared__ float Bsf[32][68];
  const int tid = threadIdx.x, lane = tid & 31, w = tid >> 5;
  const int lm = lane & 15, hi = lane >> 4;
  const int pb = blockIdx.x * 64;

  v8f acc[4] = {};
  for (int ks = 0; ks < 8; ++ks) {
    const int kb = ks * 32;
    for (int e = tid; e < 2048; e += 128) {
      int r = e >> 5, kk = e & 31;
      int pos = pb + r;
      int n = pos >> 16, ij = pos & 65535, i = ij >> 8, j = ij & 255;
      int kg = kb + kk, a = kg >> 4, b2 = kg & 15;
      As[r][kk] = (_Float16)(left[(size_t)(n * LL + i) * 16 + a] *
                             right[(size_t)(n * LL + j) * 16 + b2]);
    }
    for (int e = tid; e < 512; e += 128) {
      int r = e >> 4, c4 = (e & 15) * 4;
      async_copy_b128(&Bsf[r][c4], &gate_w[(size_t)(kb + r) * DT + c4]);
    }
    async_wait0();
    __syncthreads();
    v16h af = a_frag_f16(As, w << 4, lm, hi);
#pragma unroll
    for (int nt = 0; nt < 4; ++nt) {
      v16h bf = b_frag_f32_col<68>(Bsf, nt * 16, lm, hi);
      acc[nt] = __builtin_amdgcn_wmma_f32_16x16x32_f16(
          false, af, false, bf, (short)0, acc[nt], false, false);
    }
    __syncthreads();
  }
#pragma unroll
  for (int nt = 0; nt < 4; ++nt) {
#pragma unroll
    for (int r = 0; r < 8; ++r) {
      int pos = pb + (w << 4) + (hi << 3) + r;
      int col = nt * 16 + lm;
      float vv = acc[nt][r] + gate_b[col];
      vv = 1.f / (1.f + __expf(-vv));
      size_t o = (size_t)pos * DT + col;
      biasOut[o] = vv * rbfW[o];
    }
  }
}

// ---------------------------------------------------------------------------
// Elementwise / reduction helpers
// ---------------------------------------------------------------------------
__global__ __launch_bounds__(128) void k_layernorm(
    const float* __restrict__ x, const float* __restrict__ g,
    const float* __restrict__ b, float* __restrict__ y, int rows, int D)
{
  int wid = blockIdx.x * 4 + (threadIdx.x >> 5);
  int lane = threadIdx.x & 31;
  if (wid >= rows) return;
  const float* xr = x + (size_t)wid * D;
  float s = 0.f, sq = 0.f;
  for (int c = lane; c < D; c += 32) { float v = xr[c]; s += v; sq += v * v; }
  for (int m = 16; m >= 1; m >>= 1) {
    s += __shfl_xor(s, m, 32); sq += __shfl_xor(sq, m, 32);
  }
  float mean = s / D;
  float var = sq / D - mean * mean;
  float rs = rsqrtf(var + 1e-5f);
  float* yr = y + (size_t)wid * D;
  for (int c = lane; c < D; c += 32) yr[c] = (xr[c] - mean) * rs * g[c] + b[c];
}

__global__ __launch_bounds__(128) void k_softmax256(float* __restrict__ S, int rows)
{
  int wid = blockIdx.x * 4 + (threadIdx.x >> 5);
  int lane = threadIdx.x & 31;
  if (wid >= rows) return;
  float* r = S + (size_t)wid * 256;
  float v[8]; float mx = -1e30f;
#pragma unroll
  for (int t = 0; t < 8; ++t) { v[t] = r[lane + t * 32]; mx = fmaxf(mx, v[t]); }
  for (int m = 16; m >= 1; m >>= 1) mx = fmaxf(mx, __shfl_xor(mx, m, 32));
  float s = 0.f;
#pragma unroll
  for (int t = 0; t < 8; ++t) { v[t] = __expf(v[t] - mx); s += v[t]; }
  for (int m = 16; m >= 1; m >>= 1) s += __shfl_xor(s, m, 32);
  float inv = 1.f / s;
#pragma unroll
  for (int t = 0; t < 8; ++t) r[lane + t * 32] = v[t] * inv;
}

__global__ void k_rbf(const float* __restrict__ xyz, const float* __restrict__ mask,
                      float* __restrict__ out)
{
  size_t pos = blockIdx.x;
  int c = threadIdx.x;
  int n = (int)(pos >> 16), ij = (int)(pos & 65535), i = ij >> 8, j = ij & 255;
  const float* xi = xyz + (size_t)(n * LL + i) * 3;
  const float* xj = xyz + (size_t)(n * LL + j) * 3;
  float dx = xi[0] - xj[0], dy = xi[1] - xj[1], dz = xi[2] - xj[2];
  float d2 = dx * dx + dy * dy + dz * dz;
  float D = sqrtf(fmaxf(d2, 1e-12f));
  float mu = 2.f + 20.f * (float)c / 63.f;
  float t = (D - mu) * (64.f / 20.f);
  out[pos * DT + c] = __expf(-t * t) * mask[pos];
}

__global__ void k_concat_emb(const float* __restrict__ t1d,
                             const float* __restrict__ t2d, float* __restrict__ out)
{
  size_t e = (size_t)blockIdx.x * blockDim.x + threadIdx.x;
  if (e >= (size_t)SZPOS * 88) return;
  size_t pos = e / 88;
  int c = (int)(e % 88);
  int n = (int)(pos >> 16), ij = (int)(pos & 65535), i = ij >> 8, j = ij & 255;
  float v;
  if (c < 44)       v = t2d[pos * 44 + c];
  else if (c < 66)  v = t1d[(size_t)(n * LL + i) * 22 + (c - 44)];
  else              v = t1d[(size_t)(n * LL + j) * 22 + (c - 66)];
  out[e] = v;
}

__global__ void k_concat_t1da(const float* __restrict__ t1d,
                              const float* __restrict__ at, float* __restrict__ out)
{
  int e = blockIdx.x * blockDim.x + threadIdx.x;
  if (e >= 1024 * 52) return;
  int r = e / 52, c = e % 52;
  out[e] = (c < 22) ? t1d[r * 22 + c] : at[r * 30 + (c - 22)];
}

__global__ void k_mul(float* __restrict__ a, const float* __restrict__ g, size_t n)
{
  size_t i = (size_t)blockIdx.x * blockDim.x + threadIdx.x;
  if (i < n) a[i] *= g[i];
}

// ---------------------------------------------------------------------------
// Torsion cross-attention core (T=4 keys per (b,l) query); tiny -> VALU.
// ---------------------------------------------------------------------------
__global__ __launch_bounds__(64) void k_tor_attn(
    const float* __restrict__ state, const float* __restrict__ t1da,
    const float* __restrict__ qw, const float* __restrict__ kw,
    const float* __restrict__ vw, float* __restrict__ o)
{
  __shared__ float qs[64], ks_[4][64], vs[4][64], lg[16], aw[16];
  int l = blockIdx.x, tid = threadIdx.x;
  float acc = 0.f;
  for (int c = 0; c < 32; ++c) acc += state[l * 32 + c] * qw[c * 64 + tid];
  qs[tid] = acc * SCALE_Q;
  for (int t = 0; t < 4; ++t) {
    const float* kv = t1da + ((size_t)t * LL + l) * 64;
    float ka = 0.f, va = 0.f;
    for (int c = 0; c < 64; ++c) {
      float x = kv[c];
      ka += x * kw[c * 64 + tid];
      va += x * vw[c * 64 + tid];
    }
    ks_[t][tid] = ka; vs[t][tid] = va;
  }
  __syncthreads();
  if (tid < 16) {
    int t = tid >> 2, h = tid & 3;
    float s = 0.f;
    for (int d = 0; d < 16; ++d) s += qs[h * 16 + d] * ks_[t][h * 16 + d];
    lg[tid] = s;
  }
  __syncthreads();
  if (tid < 4) {
    float m = -1e30f;
    for (int t = 0; t < 4; ++t) m = fmaxf(m, lg[t * 4 + tid]);
    float s = 0.f;
    for (int t = 0; t < 4; ++t) { float e = __expf(lg[t * 4 + tid] - m); aw[t * 4 + tid] = e; s += e; }
    for (int t = 0; t < 4; ++t) aw[t * 4 + tid] /= s;
  }
  __syncthreads();
  float s = 0.f;
  int h = tid >> 4;
  for (int t = 0; t < 4; ++t) s += aw[t * 4 + h] * vs[t][tid];
  o[(size_t)l * 64 + tid] = s;
}

// ---------------------------------------------------------------------------
// Pair cross-attention core: one wave per query (b,i,j); T=4 keys.
// ---------------------------------------------------------------------------
__global__ __launch_bounds__(128) void k_pair_attn_core(
    const float* __restrict__ Q, const float* __restrict__ Kb,
    const float* __restrict__ Vb, float* __restrict__ O)
{
  int qi = blockIdx.x * 4 + (threadIdx.x >> 5);
  int lane = threadIdx.x & 31;
  int t = (lane >> 2) & 3, h = lane & 3;
  float lg = 0.f;
  {
    const float* qr = Q + (size_t)qi * 64 + h * 16;
    const float* kr = Kb + ((size_t)t * 65536 + qi) * 64 + h * 16;
    for (int d = 0; d < 16; ++d) lg += qr[d] * kr[d];
  }
  float m = lg;
  m = fmaxf(m, __shfl_xor(m, 4, 32));
  m = fmaxf(m, __shfl_xor(m, 8, 32));
  float e = __expf(lg - m);
  float s = e;
  s += __shfl_xor(s, 4, 32);
  s += __shfl_xor(s, 8, 32);
  float a = e / s;
#pragma unroll
  for (int half = 0; half < 2; ++half) {
    int c = lane + half * 32;
    int hc = c >> 4;
    float accv = 0.f;
    for (int tt = 0; tt < 4; ++tt) {
      float w = __shfl(a, tt * 4 + hc, 32);
      accv += w * Vb[((size_t)tt * 65536 + qi) * 64 + c];
    }
    O[(size_t)qi * 64 + c] = accv;
  }
}

// ---------------------------------------------------------------------------
// Host orchestration
// ---------------------------------------------------------------------------
extern "C" void kernel_launch(void* const* d_in, const int* in_sizes, int n_in,
                              void* d_out, int out_size, void* d_ws, size_t ws_size,
                              hipStream_t stream)
{
  (void)in_sizes; (void)n_in; (void)out_size; (void)ws_size;

  const float* t1d    = (const float*)d_in[0];
  const float* t2d    = (const float*)d_in[1];
  const float* alphaT = (const float*)d_in[2];
  const float* xyz    = (const float*)d_in[3];
  const float* mask   = (const float*)d_in[4];
  const float* pair   = (const float*)d_in[5];
  const float* state  = (const float*)d_in[6];
  auto P = [&](int i) { return (const float*)d_in[i]; };

  const size_t F64 = (size_t)SZPOS * 64;
  float* W       = (float*)d_ws;
  float* templ   = W;
  float* rbfbuf  = W + 1 * F64;
  float* rbfW    = W + 2 * F64;
  float* pn      = W + 3 * F64;   // also FFN hidden-norm and final templN
  float* bn      = W + 4 * F64;
  float* biasbuf = W + 5 * F64;
  float* qbuf    = W + 6 * F64;
  float* kbuf    = W + 7 * F64;
  float* vbuf    = W + 8 * F64;
  float* gbuf    = W + 9 * F64;
  float* Obuf    = W + 10 * F64;
  float* t1buf   = W + 11 * F64;             // 262144 x 128
  float* Sbuf    = W + 13 * F64;             // 4 * 4 * 256 * 256
  float* bproj   = Sbuf + 4194304;           // 262144 x 4
  float* catbuf  = W + 14 * F64;             // 262144 x 88 (reused as Kb)
  float* Kb      = catbuf;
  float* Vb      = catbuf + 23068672;
  float* Pq      = Vb + F64;
  float* Po      = Pq + 4194304;
  float* stbuf   = Po + 4194304;
  float* slnbuf  = stbuf + 32768;
  float* leftbuf = slnbuf + 32768;
  float* rightbuf= leftbuf + 16384;
  float* tcat    = rightbuf + 16384;
  float* t1da    = tcat + 53248;
  float* torO    = t1da + 65536;

  float* outPair  = (float*)d_out;
  float* outState = outPair + (size_t)65536 * 128;

  auto gemm = [&](const float* A, const float* Bm, const float* bias, float* C,
                  int M, int K, int Nd, float alpha, int act, int acc) {
    dim3 g((M + 63) / 64, (Nd + 63) / 64);
    int ntc = (g.y > 1) ? 4 : ((Nd + 15) >> 4);
    if (ntc > 4) ntc = 4;
    switch (ntc) {
      case 1: k_gemm_t<1><<<g, 128, 0, stream>>>(A, Bm, bias, C, M, K, Nd, alpha, act, acc); break;
      case 2: k_gemm_t<2><<<g, 128, 0, stream>>>(A, Bm, bias, C, M, K, Nd, alpha, act, acc); break;
      case 3: k_gemm_t<3><<<g, 128, 0, stream>>>(A, Bm, bias, C, M, K, Nd, alpha, act, acc); break;
      default: k_gemm_t<4><<<g, 128, 0, stream>>>(A, Bm, bias, C, M, K, Nd, alpha, act, acc); break;
    }
  };
  auto ln = [&](const float* x, const float* g, const float* b, float* y,
                int rows, int D) {
    k_layernorm<<<(rows + 3) / 4, 128, 0, stream>>>(x, g, b, y, rows, D);
  };

  // ---- template pair embedding: concat(t2d,left,right) @ emb_w + emb_b ----
  k_concat_emb<<<(unsigned)(((size_t)SZPOS * 88 + 255) / 256), 256, 0, stream>>>(t1d, t2d, catbuf);
  gemm(catbuf, P(7), P(8), templ, SZPOS, 88, 64, 1.f, 0, 0);

  // ---- rbf features and state projection ----
  k_rbf<<<SZPOS, 64, 0, stream>>>(xyz, mask, rbfbuf);
  gemm(t1d, P(9), P(10), stbuf, NTPL * LL, 22, 32, 1.f, 0, 0);

  // ---- TemplatePairStack: 2 blocks ----
  for (int blk = 0; blk < 2; ++blk) {
    const int bb = 15 + 40 * blk;
    ln(stbuf, P(bb + 0), P(bb + 1), slnbuf, NTPL * LL, 32);
    gemm(slnbuf, P(bb + 2), P(bb + 3), leftbuf, NTPL * LL, 32, 16, 1.f, 0, 0);
    gemm(slnbuf, P(bb + 4), P(bb + 5), rightbuf, NTPL * LL, 32, 16, 1.f, 0, 0);
    gemm(rbfbuf, P(bb + 8), P(bb + 9), rbfW, SZPOS, 64, 64, 1.f, 0, 0);
    k_gate_bias<<<SZPOS / 64, 128, 0, stream>>>(leftbuf, rightbuf, P(bb + 6),
                                                P(bb + 7), rbfW, biasbuf);

    // row-axial then col-axial (transpose folded into index math)
    for (int pass = 0; pass < 2; ++pass) {
      const int ab = bb + 10 + 12 * pass;     // +10 row params, +22 col params
      const int row_mode = (pass == 0) ? 1 : 0;
      ln(templ,   P(ab + 0), P(ab + 1), pn, SZPOS, 64);
      ln(biasbuf, P(ab + 2), P(ab + 3), bn, SZPOS, 64);
      gemm(pn, P(ab + 4), nullptr,   qbuf, SZPOS, 64, 64, SCALE_Q, 0, 0);
      gemm(pn, P(ab + 5), nullptr,   kbuf, SZPOS, 64, 64, 1.f / 256.f, 0, 0);
      gemm(pn, P(ab + 6), nullptr,   vbuf, SZPOS, 64, 64, 1.f, 0, 0);
      gemm(pn, P(ab + 8), P(ab + 9), gbuf, SZPOS, 64, 64, 1.f, 2, 0);
      gemm(bn, P(ab + 7), nullptr,   bproj, SZPOS, 64, 4, 1.f, 0, 0);
      k_attn_logits<<<dim3(4, 4, 16), 128, 0, stream>>>(qbuf, kbuf, bproj, Sbuf, row_mode);
      k_softmax256<<<(NTPL * 4 * LL) / 4, 128, 0, stream>>>(Sbuf, NTPL * 4 * LL);
      k_attn_out<<<dim3(4, 256, 16), 128, 0, stream>>>(Sbuf, vbuf, Obuf, row_mode);
      k_mul<<<(unsigned)((F64 + 255) / 256), 256, 0, stream>>>(Obuf, gbuf, F64);
      gemm(Obuf, P(ab + 10), P(ab + 11), templ, SZPOS, 64, 64, 1.f, 0, 1);
    }

    // FFN: templ += relu(ln(templ)@ff1 + b1) @ ff2 + b2
    ln(templ, P(bb + 34), P(bb + 35), pn, SZPOS, 64);
    gemm(pn, P(bb + 36), P(bb + 37), t1buf, SZPOS, 64, 128, 1.f, 1, 0);
    gemm(t1buf, P(bb + 38), P(bb + 39), templ, SZPOS, 128, 64, 1.f, 0, 1);
  }

  // ---- stack norm ----
  ln(templ, P(11), P(12), pn, SZPOS, 64);   // pn := templN

  // ---- torsion cross-attention: state += attn(state, t1da) ----
  k_concat_t1da<<<(1024 * 52 + 255) / 256, 256, 0, stream>>>(t1d, alphaT, tcat);
  gemm(tcat, P(13), P(14), t1da, NTPL * LL, 52, 64, 1.f, 0, 0);
  hipMemcpyAsync(outState, state, (size_t)256 * 32 * sizeof(float),
                 hipMemcpyDeviceToDevice, stream);
  k_tor_attn<<<256, 64, 0, stream>>>(state, t1da, P(100), P(101), P(102), torO);
  gemm(torO, P(103), P(104), outState, 256, 64, 32, 1.f, 0, 1);

  // ---- pair cross-attention: pair += attn(pair, templN over T) ----
  hipMemcpyAsync(outPair, pair, (size_t)65536 * 128 * sizeof(float),
                 hipMemcpyDeviceToDevice, stream);
  gemm(pair, P(95), nullptr, Pq, 65536, 128, 64, SCALE_Q, 0, 0);
  gemm(pn,   P(96), nullptr, Kb, SZPOS, 64, 64, 1.f, 0, 0);
  gemm(pn,   P(97), nullptr, Vb, SZPOS, 64, 64, 1.f, 0, 0);
  k_pair_attn_core<<<65536 / 4, 128, 0, stream>>>(Pq, Kb, Vb, Po);
  gemm(Po, P(98), P(99), outPair, 65536, 64, 128, 1.f, 0, 1);
}
